// Attention_76106820485292
// MI455X (gfx1250) — compile-verified, matrix operand order checked by reading, not run on hardware
//
#include <hip/hip_runtime.h>
#include <hip/hip_bf16.h>
#include <stdint.h>

// Problem sizes (fixed by reference)
static constexpr int BB  = 8;
static constexpr int SQn = 1024;
static constexpr int SKn = 1024;
static constexpr int HH  = 512;
static constexpr int NHn = 4;
static constexpr int DD  = 128;   // head dim
#define MASK_FILL (-4294967295.0f) // -2^32 + 1
#define LN_EPS 1e-5f

typedef __attribute__((ext_vector_type(16))) __bf16 v16bf;
typedef __attribute__((ext_vector_type(8)))  __bf16 v8bf;
typedef __attribute__((ext_vector_type(8)))  float  v8f;

#ifndef __has_builtin
#define __has_builtin(x) 0
#endif
#if __has_builtin(__builtin_amdgcn_global_load_async_to_lds_b128) && \
    __has_builtin(__builtin_amdgcn_s_wait_asynccnt)
#define HAVE_ASYNC_LDS 1
#else
#define HAVE_ASYNC_LDS 0
#endif

#define AS1 __attribute__((address_space(1)))
#define AS3 __attribute__((address_space(3)))

#if HAVE_ASYNC_LDS
// Builtin signature (from hipcc diagnostic): param0 is `int __vector(4) __device__ *`,
// i.e. addrspace(1) pointer to a 16-byte int vector; param1 is the LDS-side pointer.
typedef int v4i_vs __attribute__((vector_size(16)));
#endif

// Build a v_wmma_f32_16x16x32_bf16 A/B fragment from a row-major f32 row.
// Per ISA layout: per-lane vector elems [0..7] = K k0..k0+7, [8..15] = K k0+16..k0+23,
// where k0 = kk + 8*(lane>=16). Same mapping for A (row=M) and B (row=N of B^T).
__device__ __forceinline__ v16bf frag_from_f32(const float* __restrict__ row, int k0) {
  const float4* p0 = reinterpret_cast<const float4*>(row + k0);
  float4 a = p0[0], b = p0[1];
  const float4* p1 = reinterpret_cast<const float4*>(row + k0 + 16);
  float4 c = p1[0], d = p1[1];
  v16bf r;
  r[0]=(__bf16)a.x;  r[1]=(__bf16)a.y;  r[2]=(__bf16)a.z;  r[3]=(__bf16)a.w;
  r[4]=(__bf16)b.x;  r[5]=(__bf16)b.y;  r[6]=(__bf16)b.z;  r[7]=(__bf16)b.w;
  r[8]=(__bf16)c.x;  r[9]=(__bf16)c.y;  r[10]=(__bf16)c.z; r[11]=(__bf16)c.w;
  r[12]=(__bf16)d.x; r[13]=(__bf16)d.y; r[14]=(__bf16)d.z; r[15]=(__bf16)d.w;
  return r;
}

__device__ __forceinline__ v16bf frag_from_bf16(const __bf16* __restrict__ row, int k0) {
  v8bf lo = *reinterpret_cast<const v8bf*>(row + k0);
  v8bf hi = *reinterpret_cast<const v8bf*>(row + k0 + 16);
  v16bf r;
#pragma unroll
  for (int i = 0; i < 8; ++i) { r[i] = lo[i]; r[8 + i] = hi[i]; }
  return r;
}

// ---------------------------------------------------------------------------
// Kernel 1: projections  y = x @ W^T  (bf16 WMMA, f32 accum)
//   z=0: K = memory@Wk^T -> Kb [b][h][sk][D]
//   z=1: V = memory@Wv^T -> Vt [b][h][d][SK]   (transposed store, contiguous s)
//   z=2: Q = decoder@Wq^T -> Qb [b][h][sq][D]
// Workgroup (256 thr = 8 waves) computes a 32x256 output tile.
// K/Q stores are staged through LDS so global writes are coalesced b128s.
// ---------------------------------------------------------------------------
__global__ __launch_bounds__(256) void proj_kernel(
    const float* __restrict__ memory, const float* __restrict__ decoder,
    const float* __restrict__ Wk, const float* __restrict__ Wv, const float* __restrict__ Wq,
    __bf16* __restrict__ Qb, __bf16* __restrict__ Kb, __bf16* __restrict__ Vt)
{
  __shared__ __align__(16) __bf16 stg[8][16][72];   // per-wave 16x64 tile (+pad)

  const int z    = blockIdx.z;
  const int tid  = threadIdx.x;
  const int w    = tid >> 5;
  const int lane = tid & 31;
  const int l    = lane & 15;
  const int hiH  = lane >> 4;

  const int m0 = blockIdx.y * 32 + (w & 1) * 16;    // output row tile base
  const int n0 = blockIdx.x * 256 + (w >> 1) * 64;  // output col tile base

  const int bI = m0 >> 10;          // batch
  const int s0 = m0 & 1023;         // sequence pos base of tile

  const float* x = (z == 2) ? decoder : memory;
  const float* W = (z == 0) ? Wk : (z == 1 ? Wv : Wq);
  const float* xrow = x + ((size_t)bI * 1024 + (s0 + l)) * HH;

  v8f acc[4];
#pragma unroll
  for (int t = 0; t < 4; ++t)
#pragma unroll
    for (int i = 0; i < 8; ++i) acc[t][i] = 0.0f;

  for (int kk = 0; kk < HH; kk += 32) {
    const int k0 = kk + 8 * hiH;
    v16bf a = frag_from_f32(xrow, k0);
#pragma unroll
    for (int t = 0; t < 4; ++t) {
      const float* wrow = W + (size_t)(n0 + t * 16 + l) * HH;  // B[k][n] = W[n][k]
      v16bf bf = frag_from_f32(wrow, k0);
      acc[t] = __builtin_amdgcn_wmma_f32_16x16x32_bf16(false, a, false, bf,
                                                       (short)0, acc[t], false, false);
    }
  }

  if (z == 1) {
    // transposed store: Vt[((b*NH+h)*D + dl)*SK + s], lane writes 8 contiguous s
#pragma unroll
    for (int t = 0; t < 4; ++t) {
      const int o = n0 + t * 16 + l;
      const int h = o >> 7, dl = o & 127;
      v8bf vs;
#pragma unroll
      for (int r = 0; r < 8; ++r) vs[r] = (__bf16)acc[t][r];
      size_t addr = (((size_t)bI * NHn + h) * DD + dl) * (size_t)SKn + (s0 + 8 * hiH);
      *reinterpret_cast<v8bf*>(Vt + addr) = vs;
    }
  } else {
    __bf16* out = (z == 0) ? Kb : Qb;
    // stage C tiles into LDS, then write coalesced 16B rows
#pragma unroll
    for (int t = 0; t < 4; ++t)
#pragma unroll
      for (int r = 0; r < 8; ++r)
        stg[w][r + 8 * hiH][t * 16 + l] = (__bf16)acc[t][r];
    __syncthreads();   // uniform branch per block (z is uniform)

    const int row  = lane >> 1;           // 16 rows, 2 lanes per row
    const int half = lane & 1;            // 32 contiguous outputs each
    const int o0   = n0 + half * 32;      // 64-aligned block => single head
    const int h    = o0 >> 7, dl = o0 & 127;
    const int s    = s0 + row;
    __bf16* dst = out + (((size_t)bI * NHn + h) * 1024 + s) * DD + dl;
    const __bf16* srcp = &stg[w][row][half * 32];
#pragma unroll
    for (int i = 0; i < 4; ++i)
      *reinterpret_cast<v8bf*>(dst + i * 8) =
          *reinterpret_cast<const v8bf*>(srcp + i * 8);
  }
}

// ---------------------------------------------------------------------------
// Kernel 2: per (b, h, 16-q-row block):
//   Q block staged to LDS (async-to-LDS path when available)
//   scores = QK^T/sqrt(D) -> mask -> LDS -> parallel f32 softmax * query_mask
//   -> attn written to d_out (+ bf16 copy in LDS) -> ctx = attn @ V
// ---------------------------------------------------------------------------
__global__ __launch_bounds__(256) void attn_kernel(
    const __bf16* __restrict__ Qb, const __bf16* __restrict__ Kb,
    const __bf16* __restrict__ Vt,
    const uint8_t* __restrict__ mask, const float* __restrict__ qmask,
    float* __restrict__ attn_out, float* __restrict__ ctx)
{
  __shared__ __align__(16) float  sc[16][SKn];         // 64 KB scores/exp
  __shared__ __align__(16) __bf16 abf[16][SKn + 16];   // 32.5 KB bf16 attn
  __shared__ __align__(16) __bf16 qlds[16][DD];        // 4 KB Q block
  __shared__ float red[256];
  __shared__ float rowval[16];

  const int qblk = blockIdx.x * 16;
  const int h    = blockIdx.y;
  const int bI   = blockIdx.z;
  const int tid  = threadIdx.x;
  const int w    = tid >> 5;
  const int lane = tid & 31;
  const int l    = lane & 15;
  const int hiH  = lane >> 4;

  const size_t headQ = ((size_t)bI * NHn + h) * SQn * DD;
  const size_t headK = ((size_t)bI * NHn + h) * SKn * DD;
  const size_t headV = ((size_t)bI * NHn + h) * DD * (size_t)SKn;

  // prefetch the 16x1024 mask block while we do the score GEMM
  __builtin_prefetch(mask + ((size_t)bI * SQn + qblk + (tid >> 4)) * (size_t)SKn +
                         (tid & 15) * 64, 0, 1);

  // ---- stage Q block (16x128 bf16 = 4 KB) into LDS, shared by all 8 waves ----
  {
    const size_t qbase = headQ + (size_t)qblk * DD;  // 2048 bf16 elems
#if HAVE_ASYNC_LDS
    AS1 v4i_vs* g =
        (AS1 v4i_vs*)(uintptr_t)(const void*)(Qb + qbase + (size_t)tid * 8);
    AS3 v4i_vs* lp =
        (AS3 v4i_vs*)(uint32_t)(uintptr_t)(void*)(&qlds[0][0] + tid * 8);
    __builtin_amdgcn_global_load_async_to_lds_b128(g, lp, 0, 0);
    __builtin_amdgcn_s_wait_asynccnt(0);
#else
    reinterpret_cast<v8bf*>(&qlds[0][0])[tid] =
        *reinterpret_cast<const v8bf*>(Qb + qbase + (size_t)tid * 8);
#endif
  }
  __syncthreads();

  // ---- Phase A: scores 16x1024, wave w owns 128 columns (8 tiles) ----
  const int wcol = w * 128;
  v8f acc[8];
#pragma unroll
  for (int t = 0; t < 8; ++t)
#pragma unroll
    for (int i = 0; i < 8; ++i) acc[t][i] = 0.0f;

  for (int kk = 0; kk < DD; kk += 32) {
    const int k0 = kk + 8 * hiH;
    v16bf a = frag_from_bf16(&qlds[l][0], k0);
#pragma unroll
    for (int t = 0; t < 8; ++t) {
      const __bf16* krow = Kb + headK + (size_t)(wcol + t * 16 + l) * DD;
      v16bf bf = frag_from_bf16(krow, k0);
      acc[t] = __builtin_amdgcn_wmma_f32_16x16x32_bf16(false, a, false, bf,
                                                       (short)0, acc[t], false, false);
    }
  }

  // prefetch this wave's V rows for Phase C
  const __bf16* vrow = Vt + headV + (size_t)(w * 16 + l) * SKn;
  __builtin_prefetch(vrow, 0, 1);
  __builtin_prefetch(vrow + 512, 0, 1);
  __builtin_prefetch(vrow + 1024, 0, 1);

  const float scale = 0.08838834764831845f;  // 1/sqrt(128)
#pragma unroll
  for (int t = 0; t < 8; ++t) {
    const int col = wcol + t * 16 + l;
#pragma unroll
    for (int r = 0; r < 8; ++r) {
      const int qr = r + 8 * hiH;
      const uint8_t mb = mask[((size_t)bI * SQn + qblk + qr) * (size_t)SKn + col];
      sc[qr][col] = mb ? MASK_FILL : acc[t][r] * scale;
    }
  }
  __syncthreads();

  // ---- Phase B: all 16 rows softmaxed in parallel (16 threads per row) ----
  const int rr = tid >> 4;   // row 0..15
  const int cc = tid & 15;   // column group

  float lm = -3.4e38f;
  for (int c = cc; c < SKn; c += 16) lm = fmaxf(lm, sc[rr][c]);
  red[tid] = lm;
  __syncthreads();
  if (cc == 0) {
    float m = red[tid];
#pragma unroll
    for (int j = 1; j < 16; ++j) m = fmaxf(m, red[tid + j]);
    rowval[rr] = m;
  }
  __syncthreads();
  const float rmax = rowval[rr];

  float ls = 0.0f;
  for (int c = cc; c < SKn; c += 16) {
    float e = __expf(sc[rr][c] - rmax);
    sc[rr][c] = e;
    ls += e;
  }
  red[tid] = ls;
  __syncthreads();
  if (cc == 0) {
    float s = red[tid];
#pragma unroll
    for (int j = 1; j < 16; ++j) s += red[tid + j];
    rowval[rr] = qmask[(size_t)bI * SQn + qblk + rr] / s;
  }
  __syncthreads();
  const float inv = rowval[rr];

  const size_t attnBase = (((size_t)h * BB + bI) * SQn + qblk) * (size_t)SKn;
  for (int c = cc; c < SKn; c += 16) {
    float av = sc[rr][c] * inv;
    attn_out[attnBase + (size_t)rr * SKn + c] = av;
    abf[rr][c] = (__bf16)av;
  }
  __syncthreads();

  // ---- Phase C: ctx(16xD) = attn(16x1024) @ V_h(1024xD); wave w owns 16 d ----
  const int dbase = w * 16;
  v8f oacc;
#pragma unroll
  for (int i = 0; i < 8; ++i) oacc[i] = 0.0f;

  for (int kk = 0; kk < SKn; kk += 32) {
    const int k0 = kk + 8 * hiH;
    v16bf a = frag_from_bf16(&abf[l][0], k0);
    v16bf bf = frag_from_bf16(vrow, k0);   // vrow = Vt row (dbase+l)
    oacc = __builtin_amdgcn_wmma_f32_16x16x32_bf16(false, a, false, bf,
                                                   (short)0, oacc, false, false);
  }
#pragma unroll
  for (int r = 0; r < 8; ++r) {
    const int q = qblk + r + 8 * hiH;
    ctx[((size_t)bI * SQn + q) * HH + h * DD + dbase + l] = oacc[r];
  }
}

// ---------------------------------------------------------------------------
// Kernel 3: residual + LayerNorm, one 256-thread block per (b,q) row (H=512)
// ---------------------------------------------------------------------------
__global__ __launch_bounds__(256) void ln_kernel(
    const float* __restrict__ ctx, const float* __restrict__ dec,
    const float* __restrict__ gamma, const float* __restrict__ beta,
    float* __restrict__ outp)
{
  __shared__ float red[256];
  const int row = blockIdx.x;
  const int tid = threadIdx.x;
  const size_t base = (size_t)row * HH;

  const float x0 = ctx[base + tid] + dec[base + tid];
  const float x1 = ctx[base + tid + 256] + dec[base + tid + 256];

  red[tid] = x0 + x1; __syncthreads();
  for (int s2 = 128; s2 > 0; s2 >>= 1) {
    if (tid < s2) red[tid] += red[tid + s2];
    __syncthreads();
  }
  const float mu = red[0] * (1.0f / HH); __syncthreads();

  const float d0 = x0 - mu, d1 = x1 - mu;
  red[tid] = d0 * d0 + d1 * d1; __syncthreads();
  for (int s2 = 128; s2 > 0; s2 >>= 1) {
    if (tid < s2) red[tid] += red[tid + s2];
    __syncthreads();
  }
  const float inv = rsqrtf(red[0] * (1.0f / HH) + LN_EPS);

  outp[base + tid]       = d0 * inv * gamma[tid]       + beta[tid];
  outp[base + tid + 256] = d1 * inv * gamma[tid + 256] + beta[tid + 256];
}

// ---------------------------------------------------------------------------
extern "C" void kernel_launch(void* const* d_in, const int* in_sizes, int n_in,
                              void* d_out, int out_size, void* d_ws, size_t ws_size,
                              hipStream_t stream)
{
  const float*   memory  = (const float*)d_in[0];
  const float*   decoder = (const float*)d_in[1];
  const uint8_t* mask    = (const uint8_t*)d_in[2];   // jax bool -> 1 byte/elem
  const float*   qmask   = (const float*)d_in[3];
  const float*   Wk      = (const float*)d_in[4];
  const float*   Wv      = (const float*)d_in[5];
  const float*   Wq      = (const float*)d_in[6];
  const float*   gamma   = (const float*)d_in[7];
  const float*   beta    = (const float*)d_in[8];

  char* ws = (char*)d_ws;
  __bf16* Qb  = (__bf16*)(ws);                        //  8 MB
  __bf16* Kb  = (__bf16*)(ws + (size_t)(8u << 20));   //  8 MB
  __bf16* Vt  = (__bf16*)(ws + (size_t)(16u << 20));  //  8 MB
  float*  ctx = (float*) (ws + (size_t)(24u << 20));  // 16 MB

  float* result = (float*)d_out;                           // [B,SQ,H]
  float* attn   = result + (size_t)BB * SQn * HH;          // [NH*B,SQ,SK]

  proj_kernel<<<dim3(2, 256, 3), 256, 0, stream>>>(memory, decoder, Wk, Wv, Wq,
                                                   Qb, Kb, Vt);
  attn_kernel<<<dim3(SQn / 16, NHn, BB), 256, 0, stream>>>(Qb, Kb, Vt, mask, qmask,
                                                           attn, ctx);
  ln_kernel<<<dim3(BB * SQn), 256, 0, stream>>>(ctx, decoder, gamma, beta, result);
}
